// InterpretableMultiHeadAttention_13151189861025
// MI455X (gfx1250) — matrix operand order repaired and checked by reading, not compiled, not run on hardware
//
#include <hip/hip_runtime.h>

// ---------------------------------------------------------------------------
// MI455X / gfx1250 causal MHA: bf16 WMMA, TDM + async-to-LDS double buffering.
// ---------------------------------------------------------------------------

typedef __bf16 bf16;
typedef __attribute__((ext_vector_type(16))) __bf16 v16bf;
typedef __attribute__((ext_vector_type(8)))  float  v8f;
typedef __attribute__((ext_vector_type(4)))  unsigned int tdm4u;
typedef __attribute__((ext_vector_type(4)))  int          tdm4i;
typedef __attribute__((ext_vector_type(8)))  int          tdm8i;

#define BATCH   4
#define TLEN    2048
#define DMODEL  1024
#define NHEADS  16
#define DHEAD   64
#define NQKV    2112
#define MTOT    (BATCH * TLEN)

union Frag16 { v16bf v; uint4 q[2]; };

__device__ __forceinline__ v8f wmma_bf16(v16bf a, v16bf b, v8f c) {
    return __builtin_amdgcn_wmma_f32_16x16x32_bf16(false, a, false, b,
                                                   (short)0, c, false, false);
}

// A fragment (16x32, lane = row, row-major source, leading dim `ld` elems).
__device__ __forceinline__ v16bf load_frag_a(const bf16* base, int ld, int k0, int lane) {
    const int r = lane & 15, hi = lane >> 4;
    const bf16* p = base + (size_t)r * ld + k0 + hi * 8;
    Frag16 u;
    u.q[0] = *(const uint4*)(p);
    u.q[1] = *(const uint4*)(p + 16);
    return u.v;
}

// B fragment (32x16, lane = column, source [N][K] row-major, leading dim `ld`).
__device__ __forceinline__ v16bf load_frag_b(const bf16* base, int ld, int k0, int lane) {
    const int n = lane & 15, hi = lane >> 4;
    const bf16* p = base + (size_t)n * ld + k0 + hi * 8;
    Frag16 u;
    u.q[0] = *(const uint4*)(p);
    u.q[1] = *(const uint4*)(p + 16);
    return u.v;
}

// Async 16-byte global -> LDS copy (ASYNCcnt-tracked).
__device__ __forceinline__ void async_cp16(bf16* lds, const bf16* g) {
    unsigned loff = (unsigned)(unsigned long long)lds;   // LDS offset = low 32 bits
    unsigned long long ga = (unsigned long long)g;
    asm volatile("global_load_async_to_lds_b128 %0, %1, off"
                 :: "v"(loff), "v"(ga) : "memory");
}
__device__ __forceinline__ void wait_async0() {
    asm volatile("s_wait_asynccnt 0" ::: "memory");
}
__device__ __forceinline__ void wait_tensor0() {
#if __has_builtin(__builtin_amdgcn_s_wait_tensorcnt)
    __builtin_amdgcn_s_wait_tensorcnt(0);
#else
    asm volatile("s_wait_tensorcnt 0" ::: "memory");
#endif
}

#define HAVE_TDM __has_builtin(__builtin_amdgcn_tensor_load_to_lds)

#if HAVE_TDM
// TDM 2D tile load: `rows` x `cols` bf16 elements, row stride `stride` elems,
// packed row-major into LDS.  D# bitfields per CDNA5 ISA ch.8.
__device__ __forceinline__ void tdm_load_2d(void* lds, const void* g,
                                            unsigned rows, unsigned cols,
                                            unsigned stride) {
    const unsigned la = (unsigned)(unsigned long long)lds;
    const unsigned long long ga = (unsigned long long)g;
    tdm4u g0;
    g0[0] = 1u;                                            // count=1, user mode
    g0[1] = la;                                            // lds_addr
    g0[2] = (unsigned)ga;                                  // global_addr[31:0]
    g0[3] = (unsigned)((ga >> 32) & 0x01FFFFFFu) | (2u << 30); // addr[56:32], type=2
    tdm8i g1;
    g1[0] = (int)(1u << 16);                               // data_size=1 (2 bytes)
    g1[1] = (int)((stride & 0xFFFFu) << 16);               // tensor_dim0[15:0]
    g1[2] = (int)((stride >> 16) & 0xFFFFu);               // tensor_dim0[31:16]
    g1[3] = (int)(0x4000u | (cols << 16));                 // tensor_dim1=2^30; tile_dim0
    g1[4] = (int)(rows & 0xFFFFu);                         // tile_dim1 (tile_dim2=0)
    g1[5] = (int)stride;                                   // tensor_dim0_stride[31:0]
    g1[6] = 0;
    g1[7] = 0;
    tdm4i gz = {0, 0, 0, 0};
#if defined(__clang_major__) && (__clang_major__ >= 23)
    tdm8i gz8 = {0, 0, 0, 0, 0, 0, 0, 0};
    __builtin_amdgcn_tensor_load_to_lds(g0, g1, gz, gz, gz8, 0);
#else
    __builtin_amdgcn_tensor_load_to_lds(g0, g1, gz, gz, 0);
#endif
}
#endif

__device__ __forceinline__ float xmax16(float v) {
#pragma unroll
    for (int m = 1; m < 16; m <<= 1) v = fmaxf(v, __shfl_xor(v, m, 32));
    return v;
}
__device__ __forceinline__ float xsum16(float v) {
#pragma unroll
    for (int m = 1; m < 16; m <<= 1) v += __shfl_xor(v, m, 32);
    return v;
}

// ---------------------------------------------------------------------------
// Pack / layout kernels
// ---------------------------------------------------------------------------
__global__ __launch_bounds__(256) void pack_bf16(const float* __restrict__ src,
                                                 bf16* __restrict__ dst, int n) {
    int i = blockIdx.x * 256 + threadIdx.x;
    if (i < n) dst[i] = (bf16)src[i];
}

__global__ __launch_bounds__(256) void pack_transpose(const float* __restrict__ src,
                                                      bf16* __restrict__ dst,
                                                      int kd, int nd) {
    int i = blockIdx.x * 256 + threadIdx.x;
    if (i < kd * nd) {
        int n = i / kd, k = i % kd;
        dst[i] = (bf16)src[(size_t)k * nd + n];
    }
}

// ---------------------------------------------------------------------------
// QKV projection GEMM.  Block = 8 waves, tile 256(M) x 64(N), K-step 64.
// Wave tile 32x64: 16 WMMAs per step after a single batched fragment load.
// Tiles staged via TDM (waves 0/1 issue; TENSORcnt + barrier hand-off).
// ---------------------------------------------------------------------------
#define KSTEP 64
__global__ __launch_bounds__(256) void gemm_qkv(const bf16* __restrict__ Xb,
                                                const bf16* __restrict__ Wt,
                                                bf16* __restrict__ Qb,
                                                bf16* __restrict__ Kb,
                                                bf16* __restrict__ Vt) {
    __shared__ __attribute__((aligned(16))) bf16 As[2][256 * KSTEP]; // 32KB x2
    __shared__ __attribute__((aligned(16))) bf16 Bs[2][64 * KSTEP];  //  8KB x2

    const int tid = threadIdx.x;
    const int wave = tid >> 5, lane = tid & 31;
    const int hi = lane >> 4;
    const int mbase = blockIdx.x * 256;
    const int n0 = blockIdx.y * 64;

#if HAVE_TDM
    auto issue = [&](int k0, int buf) {
        if (wave == 0)
            tdm_load_2d(&As[buf][0], Xb + (size_t)mbase * DMODEL + k0, 256, KSTEP, DMODEL);
        if (wave == 1)
            tdm_load_2d(&Bs[buf][0], Wt + (size_t)n0 * DMODEL + k0, 64, KSTEP, DMODEL);
    };
    auto wait_fill = [&]() { if (wave < 2) wait_tensor0(); };
#else
    auto issue = [&](int k0, int buf) {
#pragma unroll
        for (int j = 0; j < 8; ++j) {               // A: 2048 16B chunks
            int c = tid + j * 256;
            int row = c >> 3, off = (c & 7) * 8;
            async_cp16(&As[buf][row * KSTEP + off],
                       Xb + (size_t)(mbase + row) * DMODEL + k0 + off);
        }
#pragma unroll
        for (int j = 0; j < 2; ++j) {               // B: 512 chunks
            int c = tid + j * 256;
            int row = c >> 3, off = (c & 7) * 8;
            async_cp16(&Bs[buf][row * KSTEP + off],
                       Wt + (size_t)(n0 + row) * DMODEL + k0 + off);
        }
    };
    auto wait_fill = [&]() { wait_async0(); };
#endif

    v8f acc[2][4] = {};
    issue(0, 0);
    int buf = 0;
    for (int k0 = 0; k0 < DMODEL; k0 += KSTEP, buf ^= 1) {
        wait_fill();
        __syncthreads();
        if (k0 + KSTEP < DMODEL) issue(k0 + KSTEP, buf ^ 1);

        // batched fragment loads, then 16 back-to-back WMMAs
        const bf16* Aw = &As[buf][(wave * 32) * KSTEP];
        v16bf a00 = load_frag_a(Aw, KSTEP, 0, lane);
        v16bf a01 = load_frag_a(Aw, KSTEP, 32, lane);
        v16bf a10 = load_frag_a(Aw + 16 * KSTEP, KSTEP, 0, lane);
        v16bf a11 = load_frag_a(Aw + 16 * KSTEP, KSTEP, 32, lane);
        v16bf bb[8];
#pragma unroll
        for (int nt = 0; nt < 4; ++nt) {
            bb[2 * nt]     = load_frag_b(&Bs[buf][nt * 16 * KSTEP], KSTEP, 0, lane);
            bb[2 * nt + 1] = load_frag_b(&Bs[buf][nt * 16 * KSTEP], KSTEP, 32, lane);
        }
#pragma unroll
        for (int nt = 0; nt < 4; ++nt) {
            acc[0][nt] = wmma_bf16(a00, bb[2 * nt],     acc[0][nt]);
            acc[0][nt] = wmma_bf16(a01, bb[2 * nt + 1], acc[0][nt]);
            acc[1][nt] = wmma_bf16(a10, bb[2 * nt],     acc[1][nt]);
            acc[1][nt] = wmma_bf16(a11, bb[2 * nt + 1], acc[1][nt]);
        }
        __syncthreads();
    }

    const int bidx = mbase >> 11;
    const int t0 = mbase & (TLEN - 1);
#pragma unroll
    for (int ms = 0; ms < 2; ++ms) {
#pragma unroll
        for (int nt = 0; nt < 4; ++nt) {
            const int nb = n0 + nt * 16 + (lane & 15);
#pragma unroll
            for (int i = 0; i < 8; ++i) {
                const int t = t0 + wave * 32 + ms * 16 + i + 8 * hi;
                const float v = acc[ms][nt][i];
                if (nb < 1024) {        // Q, pre-scaled by dh^-0.5
                    const int h = nb >> 6, d = nb & 63;
                    Qb[((size_t)((bidx * NHEADS + h) * TLEN + t)) * DHEAD + d] = (bf16)(v * 0.125f);
                } else if (nb < 2048) { // K
                    const int h = (nb - 1024) >> 6, d = nb & 63;
                    Kb[((size_t)((bidx * NHEADS + h) * TLEN + t)) * DHEAD + d] = (bf16)v;
                } else {                // V transposed [b][d][t]
                    const int d = nb - 2048;
                    Vt[((size_t)(bidx * DHEAD + d)) * TLEN + t] = (bf16)v;
                }
            }
        }
    }
}

// ---------------------------------------------------------------------------
// Flash attention: block = 4 waves x 16 query rows, 64-key tiles.
// ntile = blockIdx.x+1 is exactly the causal need for every wave in the block.
// K / V^T tiles double-buffered via per-lane async-to-LDS copies.
// ---------------------------------------------------------------------------
__global__ __launch_bounds__(128) void flash_attn(const bf16* __restrict__ Qb,
                                                  const bf16* __restrict__ Kb,
                                                  const bf16* __restrict__ Vt,
                                                  float* __restrict__ Ohead) {
    __shared__ __attribute__((aligned(16))) bf16 Ks[2][64 * 64]; // [key][d] 8KB x2
    __shared__ __attribute__((aligned(16))) bf16 Vs[2][64 * 64]; // [d][key] 8KB x2
    __shared__ __attribute__((aligned(32))) bf16 Pl[4][16 * 64]; // 8KB

    const int tid = threadIdx.x;
    const int wave = tid >> 5, lane = tid & 31;
    const int r = lane & 15, hi = lane >> 4;
    const int q0 = blockIdx.x * 64 + wave * 16;
    const int bh = blockIdx.y;
    const int bb = bh >> 4;

    const bf16* Qh = Qb + (size_t)bh * TLEN * DHEAD;
    const bf16* Kh = Kb + (size_t)bh * TLEN * DHEAD;
    const bf16* Vh = Vt + (size_t)bb * DHEAD * TLEN;
    bf16* Pw = &Pl[wave][0];

    auto issue_kv = [&](int jt, int buf) {
        const int kbase = jt * 64;
#pragma unroll
        for (int j = 0; j < 4; ++j) {               // K tile: 512 chunks
            int c = tid + j * 128;
            int row = c >> 3, off = (c & 7) * 8;
            async_cp16(&Ks[buf][row * 64 + off],
                       Kh + (size_t)(kbase + row) * DHEAD + off);
        }
#pragma unroll
        for (int j = 0; j < 4; ++j) {               // V^T tile: 512 chunks
            int c = tid + j * 128;
            int row = c >> 3, off = (c & 7) * 8;
            async_cp16(&Vs[buf][row * 64 + off],
                       Vh + (size_t)row * TLEN + kbase + off);
        }
    };

    const v16bf qa0 = load_frag_a(Qh + (size_t)q0 * DHEAD, DHEAD, 0, lane);
    const v16bf qa1 = load_frag_a(Qh + (size_t)q0 * DHEAD, DHEAD, 32, lane);

    v8f o[4] = {};
    float mr[8], lr[8];
#pragma unroll
    for (int i = 0; i < 8; ++i) { mr[i] = -3.0e38f; lr[i] = 0.0f; }

    const int ntile = blockIdx.x + 1;
    issue_kv(0, 0);
    int buf = 0;
    for (int jt = 0; jt < ntile; ++jt, buf ^= 1) {
        const int kbase = jt * 64;
        wait_async0();
        __syncthreads();
        if (jt + 1 < ntile) issue_kv(jt + 1, buf ^ 1);

        // load all 8 K fragments, then 8 back-to-back score WMMAs
        v16bf kbf[8];
#pragma unroll
        for (int st = 0; st < 4; ++st) {
            kbf[2 * st]     = load_frag_b(&Ks[buf][st * 16 * 64], 64, 0, lane);
            kbf[2 * st + 1] = load_frag_b(&Ks[buf][st * 16 * 64], 64, 32, lane);
        }
        v8f s[4] = {};
#pragma unroll
        for (int st = 0; st < 4; ++st) {
            s[st] = wmma_bf16(qa0, kbf[2 * st],     s[st]);
            s[st] = wmma_bf16(qa1, kbf[2 * st + 1], s[st]);
        }

        const bool needmask = (kbase + 63 > q0);
#pragma unroll
        for (int i = 0; i < 8; ++i) {
            float e[4];
            float rm = -3.0e38f;
#pragma unroll
            for (int st = 0; st < 4; ++st) {
                float v = s[st][i];
                if (needmask && (kbase + st * 16 + r > q0 + i + 8 * hi)) v = -3.0e38f;
                e[st] = v;
                rm = fmaxf(rm, v);
            }
            rm = xmax16(rm);
            float nm = fmaxf(mr[i], rm);
            float corr = __expf(mr[i] - nm);
            float rs = 0.0f;
#pragma unroll
            for (int st = 0; st < 4; ++st) { e[st] = __expf(e[st] - nm); rs += e[st]; }
            rs = xsum16(rs);
            lr[i] = lr[i] * corr + rs;
            mr[i] = nm;
            o[0][i] *= corr; o[1][i] *= corr; o[2][i] *= corr; o[3][i] *= corr;
#pragma unroll
            for (int st = 0; st < 4; ++st)
                Pw[(i + 8 * hi) * 64 + st * 16 + r] = (bf16)e[st];
        }
        asm volatile("s_wait_dscnt 0" ::: "memory");

        const v16bf pa0 = load_frag_a(Pw, 64, 0, lane);
        const v16bf pa1 = load_frag_a(Pw, 64, 32, lane);
#pragma unroll
        for (int dt = 0; dt < 4; ++dt) {
            v16bf vb0 = load_frag_b(&Vs[buf][dt * 16 * 64], 64, 0, lane);
            v16bf vb1 = load_frag_b(&Vs[buf][dt * 16 * 64], 64, 32, lane);
            o[dt] = wmma_bf16(pa0, vb0, o[dt]);
            o[dt] = wmma_bf16(pa1, vb1, o[dt]);
        }
        __syncthreads();
    }

    float* Oh = Ohead + (size_t)bh * TLEN * DHEAD;
#pragma unroll
    for (int i = 0; i < 8; ++i) {
        const float inv = 1.0f / lr[i];
        const int row = q0 + i + 8 * hi;
#pragma unroll
        for (int dt = 0; dt < 4; ++dt)
            Oh[(size_t)row * DHEAD + dt * 16 + r] = o[dt][i] * inv;
    }
}

// ---------------------------------------------------------------------------
__global__ __launch_bounds__(256) void mean_heads(const float* __restrict__ Ohead,
                                                  bf16* __restrict__ Am) {
    int idx = blockIdx.x * 256 + threadIdx.x;
    const int tot = BATCH * TLEN * DHEAD;
    if (idx >= tot) return;
    const int b = idx / (TLEN * DHEAD);
    const int td = idx % (TLEN * DHEAD);
    float s = 0.0f;
#pragma unroll
    for (int h = 0; h < NHEADS; ++h)
        s += Ohead[((size_t)(b * NHEADS + h)) * TLEN * DHEAD + td];
    Am[idx] = (bf16)(s * (1.0f / NHEADS));
}

// ---------------------------------------------------------------------------
__global__ __launch_bounds__(256) void gemm_out(const bf16* __restrict__ Am,
                                                const bf16* __restrict__ Wob,
                                                float* __restrict__ out) {
    const int wave = threadIdx.x >> 5, lane = threadIdx.x & 31;
    const int r = lane & 15, hi = lane >> 4;
    const int m0 = blockIdx.x * 128 + wave * 16;
    const int n0 = blockIdx.y * 64;

    const v16bf a0 = load_frag_a(Am + (size_t)m0 * DHEAD, DHEAD, 0, lane);
    const v16bf a1 = load_frag_a(Am + (size_t)m0 * DHEAD, DHEAD, 32, lane);
    v16bf bb[8];
#pragma unroll
    for (int nt = 0; nt < 4; ++nt) {
        bb[2 * nt]     = load_frag_b(Wob + (size_t)(n0 + nt * 16) * DHEAD, DHEAD, 0, lane);
        bb[2 * nt + 1] = load_frag_b(Wob + (size_t)(n0 + nt * 16) * DHEAD, DHEAD, 32, lane);
    }
#pragma unroll
    for (int nt = 0; nt < 4; ++nt) {
        v8f acc = {};
        acc = wmma_bf16(a0, bb[2 * nt],     acc);
        acc = wmma_bf16(a1, bb[2 * nt + 1], acc);
#pragma unroll
        for (int i = 0; i < 8; ++i)
            out[(size_t)(m0 + i + 8 * hi) * DMODEL + n0 + nt * 16 + r] = acc[i];
    }
}

// ---------------------------------------------------------------------------
extern "C" void kernel_launch(void* const* d_in, const int* in_sizes, int n_in,
                              void* d_out, int out_size, void* d_ws, size_t ws_size,
                              hipStream_t stream) {
    const float* x    = (const float*)d_in[0];
    const float* Wqkv = (const float*)d_in[1];
    const float* Wout = (const float*)d_in[2];
    float* out = (float*)d_out;

    char* w = (char*)d_ws;
    size_t off = 0;
    bf16* Xb  = (bf16*)(w + off); off += (size_t)MTOT * DMODEL * 2;
    bf16* Wt  = (bf16*)(w + off); off += (size_t)NQKV * DMODEL * 2;
    bf16* Qb  = (bf16*)(w + off); off += (size_t)BATCH * NHEADS * TLEN * DHEAD * 2;
    bf16* Kb  = (bf16*)(w + off); off += (size_t)BATCH * NHEADS * TLEN * DHEAD * 2;
    bf16* Vtp = (bf16*)(w + off); off += (size_t)BATCH * DHEAD * TLEN * 2;
    float* Oh = (float*)(w + off); off += (size_t)BATCH * NHEADS * TLEN * DHEAD * 4;
    bf16* Am  = (bf16*)(w + off); off += (size_t)BATCH * TLEN * DHEAD * 2;
    bf16* Wob = (bf16*)(w + off); off += (size_t)DMODEL * DHEAD * 2;

    pack_bf16<<<(MTOT * DMODEL) / 256, 256, 0, stream>>>(x, Xb, MTOT * DMODEL);
    pack_transpose<<<(NQKV * DMODEL + 255) / 256, 256, 0, stream>>>(Wqkv, Wt, DMODEL, NQKV);
    pack_transpose<<<(DMODEL * DHEAD + 255) / 256, 256, 0, stream>>>(Wout, Wob, DHEAD, DMODEL);

    gemm_qkv<<<dim3(MTOT / 256, NQKV / 64), 256, 0, stream>>>(Xb, Wt, Qb, Kb, Vtp);
    flash_attn<<<dim3(TLEN / 64, BATCH * NHEADS), 128, 0, stream>>>(Qb, Kb, Vtp, Oh);
    mean_heads<<<(BATCH * TLEN * DHEAD + 255) / 256, 256, 0, stream>>>(Oh, Am);
    gemm_out<<<dim3(MTOT / 128, DMODEL / 64), 256, 0, stream>>>(Am, Wob, out);
}
